// NystromAttention_26456998543579
// MI455X (gfx1250) — matrix operand order repaired and checked by reading, not compiled, not run on hardware
//
#include <hip/hip_runtime.h>
#include <hip/hip_bf16.h>
#include <math.h>
#include <stdint.h>

// ---------------- problem constants ----------------
#define Bc   4
#define Sc   8192
#define Dc   1024
#define Hc   16
#define HDc  64
#define Mc   64
#define SEGc 128
#define BHc  64           // B*H
#define BSc  32768        // B*S
#define SCALEc 0.125f     // 1/sqrt(64)

typedef __bf16 bf16;
typedef __attribute__((ext_vector_type(4)))  bf16  bf16x4;
typedef __attribute__((ext_vector_type(8)))  bf16  bf16x8;
typedef __attribute__((ext_vector_type(16))) bf16  bf16x16;
typedef __attribute__((ext_vector_type(4)))  float f32x4;
typedef __attribute__((ext_vector_type(8)))  float f32x8;

#define ZERO8 ((f32x8){0.f,0.f,0.f,0.f,0.f,0.f,0.f,0.f})

union BF16x16 { bf16x16 v; bf16x8 h[2]; bf16 e[16]; };

__device__ __forceinline__ f32x8 wmma_bf16(bf16x16 a, bf16x16 b, f32x8 c) {
  return __builtin_amdgcn_wmma_f32_16x16x32_bf16(false, a, false, b, (short)0, c, false, false);
}

// A-frag (16xK row-major, row = rbase+(lane&15)):
// lane<16 holds K = k0+0..7 and k0+16..23 ; lane>=16 holds K = k0+8..15 and k0+24..31
__device__ __forceinline__ bf16x16 a_frag_g(const bf16* A, size_t lda, int row, int k0, int lane) {
  int kk = (lane & 16) ? 8 : 0;
  const bf16* p = A + (size_t)row * lda + k0 + kk;
  BF16x16 u;
  u.h[0] = *(const bf16x8*)(p);
  u.h[1] = *(const bf16x8*)(p + 16);
  return u.v;
}

// B-frag (Kx16): B[k][n] = src[(n0+n)*lds + k]; lane<16 -> K k0+0..7 and k0+16..23, lane>=16 -> k0+8..15 and k0+24..31
__device__ __forceinline__ bf16x16 b_frag(const bf16* src, size_t lds_, int n0, int k0, int lane) {
  int kk = (lane & 16) ? 8 : 0;
  const bf16* p = src + (size_t)(n0 + (lane & 15)) * lds_ + k0 + kk;
  BF16x16 u;
  u.h[0] = *(const bf16x8*)(p);
  u.h[1] = *(const bf16x8*)(p + 16);
  return u.v;
}

// A-frag from LDS f32 stored column-major P[k*ldk + m]
__device__ __forceinline__ bf16x16 a_frag_lds_cm(const float* P, int ldk, int m, int k0, int lane) {
  int kk = (lane & 16) ? 8 : 0;
  BF16x16 u;
#pragma unroll
  for (int j = 0; j < 8; ++j) {
    u.e[j]     = (bf16)P[(k0 + kk + j) * ldk + m];
    u.e[j + 8] = (bf16)P[(k0 + kk + 16 + j) * ldk + m];
  }
  return u.v;
}

// A-frag from LDS f32 stored row-major P[row*ld + k]
__device__ __forceinline__ bf16x16 a_frag_lds_rm(const float* P, int ld, int row, int k0, int lane) {
  int kk = (lane & 16) ? 8 : 0;
  const float* p = P + row * ld + k0 + kk;
  BF16x16 u;
#pragma unroll
  for (int j = 0; j < 8; ++j) {
    u.e[j]     = (bf16)p[j];
    u.e[j + 8] = (bf16)p[16 + j];
  }
  return u.v;
}

// CDNA5 async global->LDS copy (128b per lane), tracked by ASYNCcnt
__device__ __forceinline__ void async_ldst_b128(unsigned lds_off, const bf16* gsrc) {
  asm volatile("global_load_async_to_lds_b128 %0, %1, off"
               :: "v"(lds_off), "v"(gsrc) : "memory");
}
__device__ __forceinline__ void wait_async0() {
  asm volatile("s_wait_asynccnt 0x0" ::: "memory");
}

// ---------------- fp32 -> bf16 convert (vectorized) ----------------
__global__ void k_cvt(const float* __restrict__ in, bf16* __restrict__ out, size_t n4) {
  size_t i = (size_t)blockIdx.x * 256 + threadIdx.x;
  if (i < n4) {
    f32x4 v = *(const f32x4*)(in + i * 4);
    bf16x4 o;
    o.x = (bf16)v.x; o.y = (bf16)v.y; o.z = (bf16)v.z; o.w = (bf16)v.w;
    *(bf16x4*)(out + i * 4) = o;
  }
}

// ---------------- WMMA GEMM: out = A @ W^T + bias ----------------
// Block tile 256x64; per wave 32x64 (8 accumulators). W slab (64x32 bf16 = 4KB)
// double-buffered in LDS via async global->LDS copies.
__global__ __launch_bounds__(256) void k_gemm(
    const bf16* __restrict__ A,
    const bf16* W0, const bf16* W1, const bf16* W2,
    const float* b0, const float* b1, const float* b2,
    bf16* o0, bf16* o1, bf16* o2, float* outf, int K)
{
  __shared__ __attribute__((aligned(64))) bf16 sW[2][64 * 32];   // [buf][n*32+k]
  const bf16*  W    = (blockIdx.z == 0) ? W0 : (blockIdx.z == 1 ? W1 : W2);
  const float* bias = (blockIdx.z == 0) ? b0 : (blockIdx.z == 1 ? b1 : b2);
  bf16*        ob   = (blockIdx.z == 0) ? o0 : (blockIdx.z == 1 ? o1 : o2);
  int t = threadIdx.x;
  int wave = t >> 5, lane = t & 31;
  int rbase = blockIdx.x * 256 + wave * 32;
  int cb    = blockIdx.y * 64;

  // async staging assignment: thread t moves 16B: row n = t>>2, 8-elem chunk (t&3)
  int sn = t >> 2, sc4 = (t & 3) * 8;
  const bf16* wsrc = W + (size_t)(cb + sn) * K + sc4;
  unsigned ldst0 = (unsigned)(uintptr_t)&sW[0][sn * 32 + sc4];
  unsigned ldst1 = (unsigned)(uintptr_t)&sW[1][sn * 32 + sc4];

  f32x8 acc[2][4];
#pragma unroll
  for (int rt = 0; rt < 2; ++rt)
#pragma unroll
    for (int tt = 0; tt < 4; ++tt) acc[rt][tt] = ZERO8;

  int ar0 = rbase + (lane & 15);

  // prologue: stage first slab
  async_ldst_b128(ldst0, wsrc);
  wait_async0();
  __syncthreads();

  int buf = 0;
  for (int k0 = 0; k0 < K; k0 += 32) {
    if (k0 + 32 < K) {   // prefetch next slab into the other buffer
      async_ldst_b128(buf ? ldst0 : ldst1, wsrc + k0 + 32);
    }
    bf16x16 a0 = a_frag_g(A, K, ar0,      k0, lane);
    bf16x16 a1 = a_frag_g(A, K, ar0 + 16, k0, lane);
    const bf16* wb = sW[buf];
#pragma unroll
    for (int tt = 0; tt < 4; ++tt) {
      bf16x16 bfr = b_frag(wb, 32, tt * 16, 0, lane);
      acc[0][tt] = wmma_bf16(a0, bfr, acc[0][tt]);
      acc[1][tt] = wmma_bf16(a1, bfr, acc[1][tt]);
    }
    wait_async0();
    __syncthreads();
    buf ^= 1;
  }

  int hi8 = (lane & 16) ? 8 : 0;
#pragma unroll
  for (int rt = 0; rt < 2; ++rt)
#pragma unroll
    for (int tt = 0; tt < 4; ++tt) {
      int col = cb + tt * 16 + (lane & 15);
      float bs = bias[col];
#pragma unroll
      for (int r = 0; r < 8; ++r) {
        int row = rbase + rt * 16 + r + hi8;
        float v = acc[rt][tt][r] + bs;
        if (outf) outf[(size_t)row * K + col] = v;
        else      ob[(size_t)row * K + col]   = (bf16)v;
      }
    }
}

// ---------------- landmarks (segment means) ----------------
__global__ void k_landmark(const bf16* __restrict__ q, const bf16* __restrict__ k,
                           bf16* __restrict__ ql, bf16* __restrict__ kl) {
  int bhm = blockIdx.x; int bh = bhm >> 6; int m = bhm & 63;
  int b = bh >> 4, h = bh & 15;
  const bf16* src = blockIdx.y ? k  : q;
  bf16*       dst = blockIdx.y ? kl : ql;
  int hd = threadIdx.x;
  const bf16* p = src + ((size_t)(b * Sc + m * SEGc)) * Dc + h * HDc + hd;
  float s = 0.f;
  for (int i = 0; i < SEGc; ++i) s += (float)p[(size_t)i * Dc];
  dst[(size_t)bh * 4096 + m * 64 + hd] = (bf16)(s * (1.0f / SEGc));
}

// ---------------- k3 row softmax stats (max & sum over S) ----------------
__global__ __launch_bounds__(256) void k_rowstats(const bf16* __restrict__ ql,
                                                  const bf16* __restrict__ kbuf,
                                                  float* __restrict__ rmax,
                                                  float* __restrict__ rsum) {
  __shared__ float sQ[64];
  __shared__ float red[256];
  int bhm = blockIdx.x; int bh = bhm >> 6; int m = bhm & 63;
  int b = bh >> 4, h = bh & 15;
  int t = threadIdx.x;
  if (t < 64) sQ[t] = (float)ql[(size_t)bh * 4096 + m * 64 + t];
  __syncthreads();
  const bf16* kh = kbuf + (size_t)b * Sc * Dc + h * HDc;
  float sc[32]; float lmax = -1e30f;
#pragma unroll 1
  for (int i = 0; i < 32; ++i) {
    int s = i * 256 + t;
    const bf16* kp = kh + (size_t)s * Dc;
    float d = 0.f;
#pragma unroll
    for (int hh = 0; hh < 8; ++hh) {
      bf16x8 kv = *(const bf16x8*)(kp + hh * 8);
#pragma unroll
      for (int j = 0; j < 8; ++j) d += sQ[hh * 8 + j] * (float)kv[j];
    }
    d *= SCALEc;
    sc[i] = d; lmax = fmaxf(lmax, d);
  }
  red[t] = lmax; __syncthreads();
  for (int off = 128; off > 0; off >>= 1) { if (t < off) red[t] = fmaxf(red[t], red[t + off]); __syncthreads(); }
  float gmax = red[0]; __syncthreads();
  float lsum = 0.f;
#pragma unroll
  for (int i = 0; i < 32; ++i) lsum += __expf(sc[i] - gmax);
  red[t] = lsum; __syncthreads();
  for (int off = 128; off > 0; off >>= 1) { if (t < off) red[t] += red[t + off]; __syncthreads(); }
  if (t == 0) { rmax[bhm] = gmax; rsum[bhm] = red[0]; }
}

// ---------------- ctx1 = softmax_s(ql . k^T) @ v  (per b,h) ----------------
__global__ __launch_bounds__(256) void k_ctx1(const bf16* __restrict__ ql,
                                              const bf16* __restrict__ kbuf,
                                              const bf16* __restrict__ vbuf,
                                              const float* __restrict__ rmax,
                                              const float* __restrict__ rsum,
                                              float* __restrict__ ctx1) {
  __shared__ __attribute__((aligned(32))) float sP[128 * 64];   // [s][m] scores -> probs
  __shared__ __attribute__((aligned(32))) bf16  sVT[64 * 128];  // [hd][s]
  __shared__ float sMax[64], sInv[64];
  int bh = blockIdx.x; int b = bh >> 4, h = bh & 15;
  int t = threadIdx.x; int wave = t >> 5, lane = t & 31;
  if (t < 64) { sMax[t] = rmax[bh * 64 + t]; sInv[t] = 1.0f / rsum[bh * 64 + t]; }
  const bf16* qlh = ql + (size_t)bh * 4096;
  const bf16* kh  = kbuf + (size_t)b * Sc * Dc + h * HDc;
  const bf16* vh  = vbuf + (size_t)b * Sc * Dc + h * HDc;
  int mt = wave >> 1; int hb = (wave & 1) * 32;
  f32x8 c0 = ZERO8, c1 = ZERO8;
  __syncthreads();
  for (int c = 0; c < Sc / 128; ++c) {
    f32x8 sa[4];
#pragma unroll
    for (int q4 = 0; q4 < 4; ++q4) sa[q4] = ZERO8;
#pragma unroll
    for (int ks = 0; ks < 2; ++ks) {
      bf16x16 bfr = b_frag(kh, Dc, c * 128 + wave * 16, ks * 32, lane);
#pragma unroll
      for (int q4 = 0; q4 < 4; ++q4) {
        bf16x16 a = a_frag_g(qlh, 64, q4 * 16 + (lane & 15), ks * 32, lane);
        sa[q4] = wmma_bf16(a, bfr, sa[q4]);
      }
    }
    int sl = wave * 16 + (lane & 15);
    int hi8 = (lane & 16) ? 8 : 0;
#pragma unroll
    for (int q4 = 0; q4 < 4; ++q4)
#pragma unroll
      for (int r = 0; r < 8; ++r)
        sP[sl * 64 + q4 * 16 + r + hi8] = sa[q4][r];
#pragma unroll 1
    for (int i = 0; i < 32; ++i) {
      int idx = i * 256 + t;
      int s_l = idx >> 6, hd = idx & 63;
      sVT[hd * 128 + s_l] = vh[(size_t)(c * 128 + s_l) * Dc + hd];
    }
    __syncthreads();
#pragma unroll 1
    for (int i = 0; i < 32; ++i) {
      int idx = i * 256 + t;
      int m = idx & 63;
      sP[idx] = __expf(sP[idx] * SCALEc - sMax[m]) * sInv[m];
    }
    __syncthreads();
#pragma unroll
    for (int ks = 0; ks < 4; ++ks) {
      bf16x16 a = a_frag_lds_cm(sP, 64, mt * 16 + (lane & 15), ks * 32, lane);
      bf16x16 bb0 = b_frag(sVT, 128, hb, ks * 32, lane);
      c0 = wmma_bf16(a, bb0, c0);
      bf16x16 bb1 = b_frag(sVT, 128, hb + 16, ks * 32, lane);
      c1 = wmma_bf16(a, bb1, c1);
    }
    __syncthreads();
  }
  int hi8o = (lane & 16) ? 8 : 0;
  float* outp = ctx1 + (size_t)bh * 4096;
#pragma unroll
  for (int r = 0; r < 8; ++r) {
    int m = mt * 16 + r + hi8o;
    outp[m * 64 + hb + (lane & 15)]      = c0[r];
    outp[m * 64 + hb + 16 + (lane & 15)] = c1[r];
  }
}

// ---------------- k2 softmax + Newton-Schulz pinv + ctx2 ----------------
__global__ __launch_bounds__(256) void k_pinv_ctx2(const bf16* __restrict__ ql,
                                                   const bf16* __restrict__ kl,
                                                   const float* __restrict__ ctx1,
                                                   bf16* __restrict__ ctx2) {
  extern __shared__ float sm[];
  float* A   = sm;            // 4096
  float* Z   = sm + 4096;
  float* T   = sm + 8192;
  float* Z2  = sm + 12288;
  float* red = sm + 16384;    // 256
  int bh = blockIdx.x; int t = threadIdx.x;
  const bf16* qlh = ql + (size_t)bh * 4096;
  const bf16* klh = kl + (size_t)bh * 4096;
#pragma unroll 1
  for (int e = 0; e < 16; ++e) {
    int idx = e * 256 + t;
    int l = idx >> 6, m = idx & 63;
    float d = 0.f;
    for (int hd = 0; hd < 64; ++hd) d += (float)qlh[l * 64 + hd] * (float)klh[m * 64 + hd];
    A[idx] = d * SCALEc;
  }
  __syncthreads();
  if (t < 64) {
    float mx = -1e30f;
    for (int j = 0; j < 64; ++j) mx = fmaxf(mx, A[t * 64 + j]);
    float s = 0.f;
    for (int j = 0; j < 64; ++j) { float e2 = __expf(A[t * 64 + j] - mx); A[t * 64 + j] = e2; s += e2; }
    float inv = 1.0f / s;
    for (int j = 0; j < 64; ++j) A[t * 64 + j] *= inv;
  }
  __syncthreads();
  float p = 0.f;
  for (int e = 0; e < 16; ++e) { float v = A[e * 256 + t]; p += v * v; }
  red[t] = p; __syncthreads();
  for (int off = 128; off > 0; off >>= 1) { if (t < off) red[t] += red[t + off]; __syncthreads(); }
  float inv_n = 1.0f / (sqrtf(red[0]) + 1e-6f);
  __syncthreads();
  for (int e = 0; e < 16; ++e) A[e * 256 + t] *= inv_n;   // An
  __syncthreads();
  for (int e = 0; e < 16; ++e) {
    int idx = e * 256 + t; int i = idx >> 6, j = idx & 63;
    Z[idx] = A[j * 64 + i];                                // Z = An^T
  }
  __syncthreads();
  for (int it = 0; it < 6; ++it) {
    for (int e = 0; e < 16; ++e) {
      int idx = e * 256 + t; int i = idx >> 6, j = idx & 63;
      float s = 0.f;
      for (int kx = 0; kx < 64; ++kx) s += A[i * 64 + kx] * Z[kx * 64 + j];
      T[idx] = s;
    }
    __syncthreads();
    for (int e = 0; e < 16; ++e) {
      int idx = e * 256 + t; int i = idx >> 6, j = idx & 63;
      float s = 0.f;
      for (int kx = 0; kx < 64; ++kx) s += Z[i * 64 + kx] * T[kx * 64 + j];
      Z2[idx] = 1.5f * Z[idx] - 0.5f * s;
    }
    __syncthreads();
    for (int e = 0; e < 16; ++e) Z[e * 256 + t] = Z2[e * 256 + t];
    __syncthreads();
  }
  const float* c1 = ctx1 + (size_t)bh * 4096;
  bf16* c2 = ctx2 + (size_t)bh * 4096;
  for (int e = 0; e < 16; ++e) {
    int idx = e * 256 + t; int l = idx >> 6, d = idx & 63;
    float s = 0.f;
    for (int m = 0; m < 64; ++m) s += Z[l * 64 + m] * c1[m * 64 + d];
    c2[idx] = (bf16)(s * inv_n);
  }
}

// ---------------- fused: out = softmax_m(q . kl^T) @ ctx2 ----------------
__global__ __launch_bounds__(256) void k_attn_out(const bf16* __restrict__ q,
                                                  const bf16* __restrict__ kl,
                                                  const bf16* __restrict__ ctx2,
                                                  bf16* __restrict__ ao) {
  __shared__ __attribute__((aligned(32))) bf16 sC2T[64 * 64];   // [hd][m]
  __shared__ float sS[8][16 * 64];                              // per-wave [s_local][m]
  int bh = blockIdx.y; int b = bh >> 4, h = bh & 15;
  int t = threadIdx.x; int wave = t >> 5, lane = t & 31;
#pragma unroll 1
  for (int e = 0; e < 16; ++e) {
    int idx = e * 256 + t;
    int m = idx >> 6, hd = idx & 63;
    sC2T[hd * 64 + m] = ctx2[(size_t)bh * 4096 + idx];
  }
  const bf16* qh  = q + (size_t)b * Sc * Dc + h * HDc;
  const bf16* klh = kl + (size_t)bh * 4096;
  int rbase = blockIdx.x * 128 + wave * 16;
  f32x8 sacc[4];
#pragma unroll
  for (int mt = 0; mt < 4; ++mt) sacc[mt] = ZERO8;
#pragma unroll
  for (int ks = 0; ks < 2; ++ks) {
    bf16x16 a = a_frag_g(qh, Dc, rbase + (lane & 15), ks * 32, lane);
#pragma unroll
    for (int mt = 0; mt < 4; ++mt) {
      bf16x16 bb = b_frag(klh, 64, mt * 16, ks * 32, lane);
      sacc[mt] = wmma_bf16(a, bb, sacc[mt]);
    }
  }
  float* P = sS[wave];
  int hi8 = (lane & 16) ? 8 : 0;
#pragma unroll
  for (int mt = 0; mt < 4; ++mt)
#pragma unroll
    for (int r = 0; r < 8; ++r)
      P[(r + hi8) * 64 + mt * 16 + (lane & 15)] = sacc[mt][r];
  __syncthreads();
  {
    int row = lane & 15;
    float* pr = P + row * 64 + (lane >> 4) * 32;
    float mx = -1e30f;
#pragma unroll
    for (int j = 0; j < 32; ++j) mx = fmaxf(mx, pr[j] * SCALEc);
    mx = fmaxf(mx, __shfl_xor(mx, 16, 32));
    float s = 0.f;
#pragma unroll
    for (int j = 0; j < 32; ++j) { float e = __expf(pr[j] * SCALEc - mx); pr[j] = e; s += e; }
    s += __shfl_xor(s, 16, 32);
    float inv = 1.0f / s;
#pragma unroll
    for (int j = 0; j < 32; ++j) pr[j] *= inv;
  }
  __syncthreads();
  f32x8 oacc[4];
#pragma unroll
  for (int ht = 0; ht < 4; ++ht) oacc[ht] = ZERO8;
#pragma unroll
  for (int ks = 0; ks < 2; ++ks) {
    bf16x16 a = a_frag_lds_rm(P, 64, lane & 15, ks * 32, lane);
#pragma unroll
    for (int ht = 0; ht < 4; ++ht) {
      bf16x16 bb = b_frag(sC2T, 64, ht * 16, ks * 32, lane);
      oacc[ht] = wmma_bf16(a, bb, oacc[ht]);
    }
  }
  bf16* aoh = ao + (size_t)b * Sc * Dc + h * HDc;
#pragma unroll
  for (int ht = 0; ht < 4; ++ht) {
    int hd = ht * 16 + (lane & 15);
#pragma unroll
    for (int r = 0; r < 8; ++r) {
      int s = rbase + r + hi8;
      aoh[(size_t)s * Dc + hd] = (bf16)(oacc[ht][r]);
    }
  }
}

// ---------------- depthwise conv residual add ----------------
__global__ __launch_bounds__(256) void k_convres(const bf16* __restrict__ v,
                                                 const float* __restrict__ cw,
                                                 const float* __restrict__ cb,
                                                 bf16* __restrict__ ao) {
  size_t idx = (size_t)blockIdx.x * 256 + threadIdx.x;
  size_t SD = (size_t)Sc * Dc;
  size_t r = idx % SD;
  int s = (int)(r / Dc); int d = (int)(r % Dc);
  float c = (float)v[idx];
  float lft = (s > 0)      ? (float)v[idx - Dc] : 0.0f;
  float rgt = (s < Sc - 1) ? (float)v[idx + Dc] : 0.0f;
  float res = lft * cw[d * 3 + 0] + c * cw[d * 3 + 1] + rgt * cw[d * 3 + 2] + cb[d];
  ao[idx] = (bf16)((float)ao[idx] + res);
}

// ---------------- host launcher ----------------
extern "C" void kernel_launch(void* const* d_in, const int* in_sizes, int n_in,
                              void* d_out, int out_size, void* d_ws, size_t ws_size,
                              hipStream_t stream) {
  (void)in_sizes; (void)n_in; (void)out_size; (void)ws_size;
  const float* x  = (const float*)d_in[0];
  const float* Wq = (const float*)d_in[1];
  const float* bq = (const float*)d_in[2];
  const float* Wk = (const float*)d_in[3];
  const float* bk = (const float*)d_in[4];
  const float* Wv = (const float*)d_in[5];
  const float* bv = (const float*)d_in[6];
  const float* Wo = (const float*)d_in[7];
  const float* bo = (const float*)d_in[8];
  const float* cw = (const float*)d_in[9];
  const float* cbv = (const float*)d_in[10];

  char* ws = (char*)d_ws;
  size_t off = 0;
  auto alloc = [&](size_t bytes) -> void* {
    void* p = ws + off; off += (bytes + 255) & ~(size_t)255; return p;
  };
  const size_t NE = (size_t)BSc * Dc;       // 33.5M elems
  const size_t WE = (size_t)Dc * Dc;        // 1M elems
  bf16* xb  = (bf16*)alloc(NE * 2);
  bf16* wqb = (bf16*)alloc(WE * 2);
  bf16* wkb = (bf16*)alloc(WE * 2);
  bf16* wvb = (bf16*)alloc(WE * 2);
  bf16* wob = (bf16*)alloc(WE * 2);
  bf16* qb  = (bf16*)alloc(NE * 2);
  bf16* kb  = (bf16*)alloc(NE * 2);
  bf16* vb  = (bf16*)alloc(NE * 2);
  bf16* qlb = (bf16*)alloc((size_t)BHc * 4096 * 2);
  bf16* klb = (bf16*)alloc((size_t)BHc * 4096 * 2);
  float* rmax = (float*)alloc((size_t)BHc * 64 * 4);
  float* rsum = (float*)alloc((size_t)BHc * 64 * 4);
  float* c1f  = (float*)alloc((size_t)BHc * 4096 * 4);
  bf16*  c2b  = (bf16*)alloc((size_t)BHc * 4096 * 2);
  bf16*  aob  = (bf16*)alloc(NE * 2);

  // 1. convert to bf16 (vectorized x4)
  k_cvt<<<dim3((unsigned)(NE / 4 / 256)), 256, 0, stream>>>(x, xb, NE / 4);
  k_cvt<<<dim3((unsigned)(WE / 4 / 256)), 256, 0, stream>>>(Wq, wqb, WE / 4);
  k_cvt<<<dim3((unsigned)(WE / 4 / 256)), 256, 0, stream>>>(Wk, wkb, WE / 4);
  k_cvt<<<dim3((unsigned)(WE / 4 / 256)), 256, 0, stream>>>(Wv, wvb, WE / 4);
  k_cvt<<<dim3((unsigned)(WE / 4 / 256)), 256, 0, stream>>>(Wo, wob, WE / 4);

  // 2. fused Q/K/V projections (WMMA + async LDS double buffering)
  k_gemm<<<dim3(BSc / 256, Dc / 64, 3), 256, 0, stream>>>(
      xb, wqb, wkb, wvb, bq, bk, bv, qb, kb, vb, nullptr, Dc);

  // 3. landmarks
  k_landmark<<<dim3(BHc * Mc, 2), 64, 0, stream>>>(qb, kb, qlb, klb);

  // 4. k3 softmax stats
  k_rowstats<<<dim3(BHc * Mc), 256, 0, stream>>>(qlb, kb, rmax, rsum);

  // 5. ctx1 (WMMA flash-style over S)
  k_ctx1<<<dim3(BHc), 256, 0, stream>>>(qlb, kb, vb, rmax, rsum, c1f);

  // 6. k2 softmax + pinv + ctx2
  k_pinv_ctx2<<<dim3(BHc), 256, (16384 + 256) * sizeof(float), stream>>>(qlb, klb, c1f, c2b);

  // 7. fused k1-softmax + out (WMMA)
  k_attn_out<<<dim3(Sc / 128, BHc), 256, 0, stream>>>(qb, klb, c2b, aob);

  // 8. depthwise conv residual
  k_convres<<<dim3((unsigned)(NE / 256)), 256, 0, stream>>>(vb, cw, cbv, aob);

  // 9. final projection to fp32 output (WMMA)
  k_gemm<<<dim3(BSc / 256, Dc / 64, 1), 256, 0, stream>>>(
      aob, wob, wob, wob, bo, bo, bo, nullptr, nullptr, nullptr, (float*)d_out, Dc);
}
